// GATv2GRUCell_78984448573529
// MI455X (gfx1250) — matrix operand order repaired and checked by reading, not compiled, not run on hardware
//
#include <hip/hip_runtime.h>
#include <hip/hip_bf16.h>
#include <math.h>

// ---------------------------------------------------------------------------
// GATv2-GRU cell for MI455X (gfx1250, wave32, WMMA).
//
// Input order (setup_inputs() insertion order, params nested dict flattened
// in insertion order u,r,c x {Wl,bl,Wr,br,We,att,bias,Wlin,blin}):
//   0:X[N*64] 1:edge_index[2*E](i32) 2:edge_attr[E*4] 3:H[N*32]
//   4+9g+0..8 : gate g params
// ---------------------------------------------------------------------------

#define IN_F   64
#define CH     32
#define HEADS  4
#define HC     128           // HEADS*CH
#define SLOPE  0.2f

typedef __attribute__((ext_vector_type(16))) _Float16 v16h;
typedef __attribute__((ext_vector_type(8)))  float    v8f;

// ---------------- float atomic max (signed, init to -inf) ------------------
__device__ __forceinline__ void atomicMaxF(float* addr, float v) {
  if (v >= 0.0f) {
    atomicMax((int*)addr, __float_as_int(v));
  } else {
    atomicMin((unsigned int*)addr, (unsigned int)__float_as_int(v));
  }
}

// ---------------- init m/den/acc per gate ----------------------------------
__global__ void init_gate(float* __restrict__ acc, float* __restrict__ m,
                          float* __restrict__ den, int n_acc, int n_seg) {
  int t = blockIdx.x * blockDim.x + threadIdx.x;
  if (t < n_acc) acc[t] = 0.0f;
  if (t < n_seg) { m[t] = -3.4e38f; den[t] = 0.0f; }
}

// ---------------- generic WMMA GEMM: C = act([A0|A1] @ B + bias) -----------
// A is a virtual [rows x 64] matrix: cols 0..31 from A0 (stride lda0),
// cols 32..63 from A1 (stride lda1).  B is [64 x NCOLT*16] row-major.
// LDS staging is FRAGMENT-MAJOR: each lane's 16-half WMMA operand is a
// contiguous 32-byte chunk -> fragment load = 2x ds_load_b128.
// act: 0 = none, 1 = sigmoid, 2 = tanh.
template <int NCOLT>
__global__ void gemm64_wmma(const float* __restrict__ A0, int lda0,
                            const float* __restrict__ A1, int lda1,
                            const float* __restrict__ B,
                            const float* __restrict__ bias,
                            float* __restrict__ C, int ldc,
                            int nrows, int act) {
  constexpr int NC = NCOLT * 16;
  // Bsf[((kk*NCOLT + wave)*32 + lane)*16 + j]
  __shared__ __align__(32) _Float16 Bsf[64 * NC];
  // Asf[(kk*32 + lane)*16 + j]
  __shared__ __align__(32) _Float16 Asf[16 * 64];

  const int tid   = threadIdx.x;
  const int bdim  = blockDim.x;          // 32 * NCOLT
  const int row0  = blockIdx.x * 16;

  // stage B (fp32 -> fp16), coalesced global read, permuted LDS write.
  // linear i over row-major B: k = i / NC, col = i % NC
  for (int i = tid; i < 64 * NC; i += bdim) {
    int k = i / NC, col = i % NC;
    int wt = col >> 4, ncol = col & 15;
    int kk = k >> 5, rem = k & 31;       // rem = half*16 + j
    int half = rem >> 4, j = rem & 15;
    int lane = half * 16 + ncol;
    Bsf[((kk * NCOLT + wt) * 32 + lane) * 16 + j] = (_Float16)B[i];
  }
  // stage 16 rows of virtual A directly in fragment-major layout.
  // i = ((kk*32 + lane)*16 + j), 1024 total
  for (int i = tid; i < 16 * 64; i += bdim) {
    int j = i & 15, lane = (i >> 4) & 31, kk = i >> 9;
    int m = lane & 15, half = lane >> 4;
    int k = kk * 32 + half * 16 + j;
    int row = row0 + m;
    float v = 0.0f;
    if (row < nrows)
      v = (k < 32) ? A0[row * lda0 + k] : A1[row * lda1 + (k - 32)];
    Asf[i] = (_Float16)v;
  }
  __syncthreads();

  const int wave  = tid >> 5;            // column tile handled by this wave
  const int lane  = tid & 31;
  const int half  = lane >> 4;
  const int ncol  = lane & 15;

  v8f acc = {};
  #pragma unroll
  for (int kk = 0; kk < 2; ++kk) {       // K = 64 = 2 x 32
    v16h a = *(const v16h*)&Asf[(kk * 32 + lane) * 16];
    v16h b = *(const v16h*)&Bsf[((kk * NCOLT + wave) * 32 + lane) * 16];
    acc = __builtin_amdgcn_wmma_f32_16x16x32_f16(
        false, a, false, b, (short)0, acc, false, false);
  }

  const int col = wave * 16 + ncol;
  const float bv = bias[col];
  #pragma unroll
  for (int i = 0; i < 8; ++i) {
    int row = row0 + i + 8 * half;
    if (row < nrows) {
      float t = acc[i] + bv;
      if (act == 1)      t = 1.0f / (1.0f + __expf(-t));
      else if (act == 2) t = tanhf(t);
      C[row * ldc + col] = t;
    }
  }
}

// ---------------- edge pass 1: logits + segment max ------------------------
// one wave per edge; lane = channel within head
__global__ void edge_logits(const float* __restrict__ XL,
                            const float* __restrict__ XR,
                            const int* __restrict__ src,
                            const int* __restrict__ dst,
                            const float* __restrict__ EA,
                            const float* __restrict__ We,
                            const float* __restrict__ att,
                            float* __restrict__ logits,
                            float* __restrict__ mbuf, int E) {
  __shared__ float Wes[4 * HC];
  __shared__ float atts[HC];
  const int tid = threadIdx.x;
  for (int i = tid; i < 4 * HC; i += blockDim.x) Wes[i] = We[i];
  for (int i = tid; i < HC;     i += blockDim.x) atts[i] = att[i];
  __syncthreads();

  const int e = blockIdx.x * (blockDim.x >> 5) + (tid >> 5);
  if (e >= E) return;
  const int lane = tid & 31;
  const int s = src[e], d = dst[e];
  const float ea0 = EA[e * 4 + 0], ea1 = EA[e * 4 + 1];
  const float ea2 = EA[e * 4 + 2], ea3 = EA[e * 4 + 3];

  float sh[HEADS];
  #pragma unroll
  for (int h = 0; h < HEADS; ++h) {
    const int idx = h * CH + lane;
    float xe = ea0 * Wes[idx] + ea1 * Wes[HC + idx] +
               ea2 * Wes[2 * HC + idx] + ea3 * Wes[3 * HC + idx];
    float z = XL[s * HC + idx] + XR[d * HC + idx] + xe;
    float lr = (z > 0.0f) ? z : SLOPE * z;
    sh[h] = lr * atts[idx];
  }
  #pragma unroll
  for (int off = 16; off > 0; off >>= 1) {
    sh[0] += __shfl_xor(sh[0], off, 32);
    sh[1] += __shfl_xor(sh[1], off, 32);
    sh[2] += __shfl_xor(sh[2], off, 32);
    sh[3] += __shfl_xor(sh[3], off, 32);
  }
  if (lane == 0) {
    #pragma unroll
    for (int h = 0; h < HEADS; ++h) {
      logits[e * HEADS + h] = sh[h];
      atomicMaxF(&mbuf[d * HEADS + h], sh[h]);
    }
  }
}

// ---------------- edge pass 2: exp(logit - m), den accumulation ------------
__global__ void edge_expden(const int* __restrict__ dst,
                            float* __restrict__ logits,   // in: logit, out: ex
                            const float* __restrict__ mbuf,
                            float* __restrict__ den, int E) {
  int t = blockIdx.x * blockDim.x + threadIdx.x;
  if (t >= E * HEADS) return;
  int e = t >> 2, h = t & 3;
  int d = dst[e];
  float ex = __expf(logits[t] - mbuf[d * HEADS + h]);
  logits[t] = ex;
  atomicAdd(&den[d * HEADS + h], ex);
}

// ---------------- edge pass 3: alpha-weighted scatter ----------------------
__global__ void edge_scatter(const float* __restrict__ XL,
                             const int* __restrict__ src,
                             const int* __restrict__ dst,
                             const float* __restrict__ ex,
                             const float* __restrict__ den,
                             float* __restrict__ acc, int E) {
  const int tid = threadIdx.x;
  const int e = blockIdx.x * (blockDim.x >> 5) + (tid >> 5);
  if (e >= E) return;
  const int lane = tid & 31;
  const int s = src[e], d = dst[e];
  float alpha[HEADS];
  #pragma unroll
  for (int h = 0; h < HEADS; ++h)
    alpha[h] = ex[e * HEADS + h] / (den[d * HEADS + h] + 1e-16f);
  #pragma unroll
  for (int h = 0; h < HEADS; ++h) {
    const int idx = h * CH + lane;
    atomicAdd(&acc[d * HC + idx], alpha[h] * XL[s * HC + idx]);
  }
}

// ---------------- head mean + conv bias ------------------------------------
__global__ void head_mean(const float* __restrict__ acc,
                          const float* __restrict__ bias,
                          float* __restrict__ out, int N) {
  int t = blockIdx.x * blockDim.x + threadIdx.x;
  if (t >= N * CH) return;
  int n = t >> 5, ch = t & 31;
  const float* a = acc + n * HC;
  out[t] = 0.25f * (a[ch] + a[CH + ch] + a[2 * CH + ch] + a[3 * CH + ch]) +
           bias[ch];
}

// ---------------- elementwise helpers --------------------------------------
__global__ void ew_mul(const float* __restrict__ a, const float* __restrict__ b,
                       float* __restrict__ o, int n) {
  int t = blockIdx.x * blockDim.x + threadIdx.x;
  if (t < n) o[t] = a[t] * b[t];
}

__global__ void ew_final(const float* __restrict__ u, const float* __restrict__ H,
                         const float* __restrict__ c, float* __restrict__ o, int n) {
  int t = blockIdx.x * blockDim.x + threadIdx.x;
  if (t < n) o[t] = u[t] * H[t] + (1.0f - u[t]) * c[t];
}

// ---------------------------------------------------------------------------
extern "C" void kernel_launch(void* const* d_in, const int* in_sizes, int n_in,
                              void* d_out, int out_size, void* d_ws, size_t ws_size,
                              hipStream_t stream) {
  const int N = in_sizes[0] / IN_F;
  const int E = in_sizes[1] / 2;

  const float* X  = (const float*)d_in[0];
  const int*   EI = (const int*)d_in[1];
  const int*   src = EI;
  const int*   dst = EI + E;
  const float* EA = (const float*)d_in[2];
  const float* H  = (const float*)d_in[3];

  // workspace layout (floats)
  float* ws = (float*)d_ws;
  size_t off = 0;
  float* XL     = ws + off; off += (size_t)N * HC;
  float* XR     = ws + off; off += (size_t)N * HC;
  float* acc    = ws + off; off += (size_t)N * HC;
  float* logits = ws + off; off += (size_t)E * HEADS;
  float* mbuf   = ws + off; off += (size_t)N * HEADS;
  float* den    = ws + off; off += (size_t)N * HEADS;
  float* cg     = ws + off; off += (size_t)N * CH;   // per-gate conv output
  float* ug     = ws + off; off += (size_t)N * CH;
  float* rg     = ws + off; off += (size_t)N * CH;
  float* rH     = ws + off; off += (size_t)N * CH;
  float* cpre   = ws + off; off += (size_t)N * CH;

  const int mt   = (N + 15) / 16;                 // M tiles
  const int blkE = (E + 7) / 8;                   // wave-per-edge blocks
  const int blkEH = (E * HEADS + 255) / 256;
  const int blkNC = (N * CH + 255) / 256;
  const int blkInit = (N * HC + 255) / 256;

  for (int g = 0; g < 3; ++g) {
    const float* Wl   = (const float*)d_in[4 + 9 * g + 0];
    const float* bl   = (const float*)d_in[4 + 9 * g + 1];
    const float* Wr   = (const float*)d_in[4 + 9 * g + 2];
    const float* br   = (const float*)d_in[4 + 9 * g + 3];
    const float* We   = (const float*)d_in[4 + 9 * g + 4];
    const float* att  = (const float*)d_in[4 + 9 * g + 5];
    const float* cb   = (const float*)d_in[4 + 9 * g + 6];
    const float* Wlin = (const float*)d_in[4 + 9 * g + 7];
    const float* blin = (const float*)d_in[4 + 9 * g + 8];

    init_gate<<<blkInit, 256, 0, stream>>>(acc, mbuf, den, N * HC, N * HEADS);

    // XL = f16wmma(X @ Wl) + bl ; XR = f16wmma(X @ Wr) + br
    gemm64_wmma<8><<<mt, 256, 0, stream>>>(X, IN_F, X + 32, IN_F, Wl, bl,
                                           XL, HC, N, /*act*/0);
    gemm64_wmma<8><<<mt, 256, 0, stream>>>(X, IN_F, X + 32, IN_F, Wr, br,
                                           XR, HC, N, /*act*/0);

    edge_logits<<<blkE, 256, 0, stream>>>(XL, XR, src, dst, EA, We, att,
                                          logits, mbuf, E);
    edge_expden<<<blkEH, 256, 0, stream>>>(dst, logits, mbuf, den, E);
    edge_scatter<<<blkE, 256, 0, stream>>>(XL, src, dst, logits, den, acc, E);

    head_mean<<<blkNC, 256, 0, stream>>>(acc, cb, cg, N);

    if (g == 0) {
      // u = sigmoid([cg | H] @ Wlin + blin)
      gemm64_wmma<2><<<mt, 64, 0, stream>>>(cg, CH, H, CH, Wlin, blin,
                                            ug, CH, N, /*act*/1);
    } else if (g == 1) {
      // r = sigmoid([cg | H] @ Wlin + blin) ; rH = r * H
      gemm64_wmma<2><<<mt, 64, 0, stream>>>(cg, CH, H, CH, Wlin, blin,
                                            rg, CH, N, /*act*/1);
      ew_mul<<<blkNC, 256, 0, stream>>>(rg, H, rH, N * CH);
    } else {
      // c = tanh([cg | r*H] @ Wlin + blin)
      gemm64_wmma<2><<<mt, 64, 0, stream>>>(cg, CH, rH, CH, Wlin, blin,
                                            cpre, CH, N, /*act*/2);
    }
  }

  ew_final<<<blkNC, 256, 0, stream>>>(ug, H, cpre, (float*)d_out, N * CH);
}
